// PytorchMD_15650860826882
// MI455X (gfx1250) — compile-verified, matrix operand order checked by reading, not run on hardware
//
#include <hip/hip_runtime.h>

typedef __attribute__((ext_vector_type(16))) _Float16 v16h;
typedef __attribute__((ext_vector_type(8)))  float    v8f;

#define NNEUR   16384
#define NUM_MD  1024
#define ALPHA   0.0125f        // DT / (0.02*4)
#define TRATE   0.001f         // 1/200/5
#define C1      2.5e-5f        // 0.5*HEBB_LR * 0.5  (s = 2*(MDoutTrace-0.5))
#define C2      2.5e-6f        // 0.1 * C1
#define MULT_HI 9.33333333f    // 7.0 / 0.75
#define LDSTRIDE 36            // 32 + pad, multiple of 4 for b128-aligned LDS rows

__device__ __forceinline__ float clampf(float v, float lo, float hi) {
  return fminf(fmaxf(v, lo), hi);
}

// ---------------------------------------------------------------------------
// Kernel 1: MDinp_new = MDinp + ALPHA*(-MDinp + W1 @ x)    (one block per row)
// float4 loads: 512B contiguous per wave-instruction.
// ---------------------------------------------------------------------------
__global__ void __launch_bounds__(256) md_matvec(const float* __restrict__ W1,
                                                 const float* __restrict__ x,
                                                 const float* __restrict__ MDinp,
                                                 float* __restrict__ mdinp_new) {
  const int row = blockIdx.x;
  const int tid = threadIdx.x;
  const float* wrow = W1 + (size_t)row * NNEUR;
  float acc = 0.0f;
  for (int j = (tid << 2); j < NNEUR; j += 1024) {
    __builtin_prefetch(wrow + j + 8192, 0, 0);   // global_prefetch_b8
    const float4 wv = *(const float4*)(wrow + j);
    const float4 xv = *(const float4*)(x + j);
    acc = fmaf(wv.x, xv.x, acc);
    acc = fmaf(wv.y, xv.y, acc);
    acc = fmaf(wv.z, xv.z, acc);
    acc = fmaf(wv.w, xv.w, acc);
  }
  // wave32 reduction
  for (int off = 16; off > 0; off >>= 1)
    acc += __shfl_down(acc, off, 32);
  __shared__ float part[8];
  const int wave = tid >> 5, lane = tid & 31;
  if (lane == 0) part[wave] = acc;
  __syncthreads();
  if (tid == 0) {
    float dot = 0.0f;
#pragma unroll
    for (int w = 0; w < 8; ++w) dot += part[w];
    const float mi = MDinp[row];
    mdinp_new[row] = mi + ALPHA * (dot - mi);
  }
}

// ---------------------------------------------------------------------------
// Kernel 2: WTA thresholds, traces, mean(pre); stash s (=±1) and b (=pre-thr)
// ---------------------------------------------------------------------------
__device__ __forceinline__ void top2_reduce(float* s1, float* s2, int tid) {
  for (int st = 512; st > 0; st >>= 1) {
    __syncthreads();
    if (tid < st) {
      const float a1 = s1[tid], a2 = s2[tid];
      const float b1 = s1[tid + st], b2 = s2[tid + st];
      if (a1 >= b1) { s1[tid] = a1; s2[tid] = fmaxf(a2, b1); }
      else          { s1[tid] = b1; s2[tid] = fmaxf(b2, a1); }
    }
  }
  __syncthreads();
}

__global__ void __launch_bounds__(1024) md_small(const float* __restrict__ x,
                                                 const float* __restrict__ preT,
                                                 const float* __restrict__ postT,
                                                 const float* __restrict__ mdinp_new,
                                                 float* __restrict__ o_mdout,
                                                 float* __restrict__ o_pre,
                                                 float* __restrict__ o_post,
                                                 float* __restrict__ ws_s,
                                                 float* __restrict__ ws_b) {
  __shared__ float s1[NUM_MD];
  __shared__ float s2[NUM_MD];
  const int tid = threadIdx.x;

  // ---- WTA(MDinp_new): threshold = mean of top-2 ----
  const float y = mdinp_new[tid];
  s1[tid] = y; s2[tid] = -3.4e38f;
  top2_reduce(s1, s2, tid);
  const float thr = 0.5f * (s1[0] + s2[0]);
  const float mo = (y >= thr) ? 1.0f : 0.0f;
  o_mdout[tid] = mo;

  // ---- post trace + WTA(post) ----
  const float pt = postT[tid];
  const float post = pt + TRATE * (mo - pt);
  o_post[tid] = post;
  __syncthreads();                // done reading s1[0]/s2[0]
  s1[tid] = post; s2[tid] = -3.4e38f;
  top2_reduce(s1, s2, tid);
  const float thr2 = 0.5f * (s1[0] + s2[0]);
  ws_s[tid] = (post >= thr2) ? 1.0f : -1.0f;   // s = 2*(MDoutTrace - 0.5)

  // ---- pre trace (16 elems/thread, stride-1024 coalesced) + mean ----
  float prev[16];
  float lsum = 0.0f;
#pragma unroll
  for (int k = 0; k < 16; ++k) {
    const int j = (k << 10) + tid;
    const float p0 = preT[j];
    const float p  = p0 + TRATE * (x[j] - p0);
    prev[k] = p;
    o_pre[j] = p;
    lsum += p;
  }
  __syncthreads();
  s1[tid] = lsum;
  for (int st = 512; st > 0; st >>= 1) {
    __syncthreads();
    if (tid < st) s1[tid] += s1[tid + st];
  }
  __syncthreads();
  const float pre_thr = s1[0] * (1.0f / (float)NNEUR);
#pragma unroll
  for (int k = 0; k < 16; ++k) {
    const int j = (k << 10) + tid;
    ws_b[j] = prev[k] - pre_thr;
  }
}

// ---------------------------------------------------------------------------
// Kernel 3: Hebbian update. One wave per 32x32 region.
//  - 4x v_wmma_f32_16x16x32_f16 compute the rank-1 delta sub-tiles
//    (s in A K=0 slot, b in B K=0 slot -> D[m][n] = s_m * b_n).
//  - Delta staged to LDS both row-major (Dbuf) and transposed (Tbuf, via
//    store addressing), stride 36 floats for b128 alignment / bank spread.
//  - All global IO is float4 (b128): 4 rows x 128B contiguous per wave-instruction.
// ---------------------------------------------------------------------------
__global__ void __launch_bounds__(256) md_hebb_update(const float* __restrict__ W1,
                                                      const float* __restrict__ W2,
                                                      const float* __restrict__ W3,
                                                      const float* __restrict__ ws_s,
                                                      const float* __restrict__ ws_b,
                                                      float* __restrict__ O1,
                                                      float* __restrict__ O2,
                                                      float* __restrict__ O3) {
  __shared__ float ldsD[8][32 * LDSTRIDE];
  __shared__ float ldsT[8][32 * LDSTRIDE];
  const int wave = threadIdx.x >> 5;
  const int lane = threadIdx.x & 31;
  const int region = blockIdx.x * 8 + wave;
  const int rj = region & 511;       // NNEUR/32 = 512 j-regions
  const int ri = region >> 9;        // NUM_MD/32 = 32 i-regions
  const int i0 = ri << 5;
  const int j0 = rj << 5;
  const int n  = lane & 15;
  const bool lo = lane < 16;
  float* Dbuf = ldsD[wave];
  float* Tbuf = ldsT[wave];

  const float s0f = ws_s[i0 + n];        // ±1, exact in f16
  const float s1f = ws_s[i0 + 16 + n];
  const float b0f = ws_b[j0 + n];        // ~1e-3, f16-normal range
  const float b1f = ws_b[j0 + 16 + n];
  const _Float16 zh = (_Float16)0.0f;

  // A (16x32 f16): lane<16 holds K=0..7 in v[0..3]; value at K=0 only.
  // B (32x16 f16): lane<16 holds K=0..7 in v[0..3]; value at K=0 only.
  v16h A0 = {}, A1 = {}, B0 = {}, B1 = {};
  A0[0] = lo ? (_Float16)s0f : zh;
  A1[0] = lo ? (_Float16)s1f : zh;
  B0[0] = lo ? (_Float16)b0f : zh;
  B1[0] = lo ? (_Float16)b1f : zh;
  const v8f cz = {};
  v8f d00 = __builtin_amdgcn_wmma_f32_16x16x32_f16(false, A0, false, B0, (short)0, cz, false, false);
  v8f d01 = __builtin_amdgcn_wmma_f32_16x16x32_f16(false, A0, false, B1, (short)0, cz, false, false);
  v8f d10 = __builtin_amdgcn_wmma_f32_16x16x32_f16(false, A1, false, B0, (short)0, cz, false, false);
  v8f d11 = __builtin_amdgcn_wmma_f32_16x16x32_f16(false, A1, false, B1, (short)0, cz, false, false);

  const int mb = lo ? 0 : 8;   // D layout: vgpr v -> M=v (lanes 0-15), M=v+8 (16-31)

  // Stage delta (row-major) and its transpose into LDS.
#define STAGE(D, II, JJ)                                                      \
  do {                                                                        \
    _Pragma("unroll")                                                         \
    for (int v = 0; v < 8; ++v)                                               \
      Dbuf[((II)*16 + v + mb) * LDSTRIDE + (JJ)*16 + n] = (D)[v];             \
    float4 t0 = make_float4((D)[0], (D)[1], (D)[2], (D)[3]);                  \
    float4 t1 = make_float4((D)[4], (D)[5], (D)[6], (D)[7]);                  \
    float* tp = Tbuf + ((JJ)*16 + n) * LDSTRIDE + (II)*16 + mb;               \
    *(float4*)tp = t0;                                                        \
    *(float4*)(tp + 4) = t1;                                                  \
  } while (0)

  STAGE(d00, 0, 0);
  STAGE(d01, 0, 1);
  STAGE(d10, 1, 0);
  STAGE(d11, 1, 1);
#undef STAGE
  __syncthreads();

  // Read-modify-write all three matrices with float4 accesses.
#pragma unroll
  for (int k = 0; k < 8; ++k) {
    const int strip = (k << 5) + lane;
    const int r  = strip >> 3;         // 0..31
    const int c  = (strip & 7) << 2;   // 0,4,...,28

    // wPFC2MD [1024, 16384]: row i0+r, cols j0+c..+3, delta row-major
    const float4 dv = *(const float4*)(Dbuf + r * LDSTRIDE + c);
    const size_t g1 = (size_t)(i0 + r) * NNEUR + (size_t)(j0 + c);
    const float4 w1 = *(const float4*)(W1 + g1);
    float4 o1;
    o1.x = clampf(fmaf(C1, dv.x, w1.x), 0.0f, 1.0f);
    o1.y = clampf(fmaf(C1, dv.y, w1.y), 0.0f, 1.0f);
    o1.z = clampf(fmaf(C1, dv.z, w1.z), 0.0f, 1.0f);
    o1.w = clampf(fmaf(C1, dv.w, w1.w), 0.0f, 1.0f);
    *(float4*)(O1 + g1) = o1;

    // wMD2PFC / wMD2PFCMult [16384, 1024]: row j0+r, cols i0+c..+3, transposed delta
    const float4 tv = *(const float4*)(Tbuf + r * LDSTRIDE + c);
    const size_t g2 = (size_t)(j0 + r) * NUM_MD + (size_t)(i0 + c);
    const float4 w2 = *(const float4*)(W2 + g2);
    float4 o2;
    o2.x = clampf(fmaf(C2, tv.x, w2.x), -10.0f, 0.0f);
    o2.y = clampf(fmaf(C2, tv.y, w2.y), -10.0f, 0.0f);
    o2.z = clampf(fmaf(C2, tv.z, w2.z), -10.0f, 0.0f);
    o2.w = clampf(fmaf(C2, tv.w, w2.w), -10.0f, 0.0f);
    *(float4*)(O2 + g2) = o2;

    const float4 w3 = *(const float4*)(W3 + g2);
    float4 o3;
    o3.x = clampf(fmaf(C2, tv.x, w3.x), 0.0f, MULT_HI);
    o3.y = clampf(fmaf(C2, tv.y, w3.y), 0.0f, MULT_HI);
    o3.z = clampf(fmaf(C2, tv.z, w3.z), 0.0f, MULT_HI);
    o3.w = clampf(fmaf(C2, tv.w, w3.w), 0.0f, MULT_HI);
    *(float4*)(O3 + g2) = o3;
  }
}

// ---------------------------------------------------------------------------
extern "C" void kernel_launch(void* const* d_in, const int* in_sizes, int n_in,
                              void* d_out, int out_size, void* d_ws, size_t ws_size,
                              hipStream_t stream) {
  const float* x     = (const float*)d_in[0];
  const float* W1    = (const float*)d_in[1];  // wPFC2MD    [1024,16384]
  const float* W2    = (const float*)d_in[2];  // wMD2PFC    [16384,1024]
  const float* W3    = (const float*)d_in[3];  // wMD2PFCMult[16384,1024]
  const float* MDinp = (const float*)d_in[4];
  const float* preT  = (const float*)d_in[5];
  const float* postT = (const float*)d_in[6];

  float* out = (float*)d_out;
  const size_t NW = (size_t)NUM_MD * (size_t)NNEUR;
  float* o_mdout = out;                  // [1024]
  float* o_w1    = o_mdout + NUM_MD;     // [1024,16384]
  float* o_w2    = o_w1 + NW;            // [16384,1024]
  float* o_w3    = o_w2 + NW;            // [16384,1024]
  float* o_mdinp = o_w3 + NW;            // [1024]
  float* o_pre   = o_mdinp + NUM_MD;     // [16384]
  float* o_post  = o_pre + NNEUR;        // [1024]

  float* ws_s = (float*)d_ws;            // [1024]  s = ±1
  float* ws_b = ws_s + NUM_MD;           // [16384] b = pre - mean(pre)

  md_matvec<<<NUM_MD, 256, 0, stream>>>(W1, x, MDinp, o_mdinp);
  md_small<<<1, 1024, 0, stream>>>(x, preT, postT, o_mdinp,
                                   o_mdout, o_pre, o_post, ws_s, ws_b);
  // (NUM_MD/32)*(NNEUR/32) = 16384 regions, 8 waves/block -> 2048 blocks
  md_hebb_update<<<2048, 256, 0, stream>>>(W1, W2, W3, ws_s, ws_b,
                                           o_w1, o_w2, o_w3);
}
